// GraphActorCritic_36550171689338
// MI455X (gfx1250) — matrix operand order repaired and checked by reading, not compiled, not run on hardware
//
#include <hip/hip_runtime.h>
#include <hip/hip_bf16.h>

typedef __attribute__((ext_vector_type(16))) _Float16 v16h;
typedef __attribute__((ext_vector_type(8)))  _Float16 v8h;
typedef __attribute__((ext_vector_type(8)))  float    v8f;

static __device__ __forceinline__ v16h pack16(v8h lo, v8h hi) {
  v16h r;
#pragma unroll
  for (int i = 0; i < 8; ++i) { r[i] = lo[i]; r[8 + i] = hi[i]; }
  return r;
}

static __device__ __forceinline__ v8f zero8() {
  v8f z;
#pragma unroll
  for (int i = 0; i < 8; ++i) z[i] = 0.0f;
  return z;
}

// ---------------- init: acc1 = 0, deg = 1 (self loop), gsum = 0 ----------------
__global__ void k_init(float* __restrict__ acc1, float* __restrict__ deg,
                       float* __restrict__ gsum, int N) {
  long long nacc = (long long)N * 128;
  long long total = nacc + N + 128;
  for (long long i = blockIdx.x * (long long)blockDim.x + threadIdx.x; i < total;
       i += (long long)gridDim.x * blockDim.x) {
    if (i < nacc)            acc1[i] = 0.0f;
    else if (i < nacc + N)   deg[i - nacc] = 1.0f;   // +1 for the self loop
    else                     gsum[i - nacc - N] = 0.0f;
  }
}

// ---------------- degree over edge rows ----------------
__global__ void k_degree(const int* __restrict__ ei, float* __restrict__ deg, int E) {
  int e = blockIdx.x * blockDim.x + threadIdx.x;
  if (e < E) atomicAdd(&deg[ei[e]], 1.0f);
}

// ---------------- deg -> rsqrt(deg) in place ----------------
__global__ void k_rsqrt(float* __restrict__ deg, int N) {
  int i = blockIdx.x * blockDim.x + threadIdx.x;
  if (i < N) {
    float d = deg[i];
    deg[i] = (d > 0.0f) ? rsqrtf(d) : 0.0f;
  }
}

// ---------------- convert lin_w2 [128x128] and aw1 [128x256] to f16 ----------------
__global__ void k_cvtw(const float* __restrict__ w2, const float* __restrict__ aw1,
                       _Float16* __restrict__ w2h, _Float16* __restrict__ aw1h) {
  int i = blockIdx.x * blockDim.x + threadIdx.x;
  if (i < 128 * 128) w2h[i] = (_Float16)w2[i];
  if (i < 128 * 256) aw1h[i] = (_Float16)aw1[i];
}

// ---------------- conv1: xw1 = node_features @ lin_w1^T + b1  (K=10) ----------------
__global__ __launch_bounds__(128) void k_xw1(const float* __restrict__ nf,
                                             const float* __restrict__ w1,
                                             const float* __restrict__ b1,
                                             float* __restrict__ out, int N) {
  __shared__ float f[10];
  int node = blockIdx.x;
  int j = threadIdx.x;
  if (node >= N) return;
  if (j < 10) f[j] = nf[node * 10 + j];
  __syncthreads();
  float s = b1[j];
#pragma unroll
  for (int k = 0; k < 10; ++k) s += w1[j * 10 + k] * f[k];
  out[node * 128 + j] = s;
}

// ---------------- msg1: acc1[col] += norm * (xw1[row] + pad(ee1)) ----------------
__global__ __launch_bounds__(128) void k_msg1(const int* __restrict__ ei,
                                              const float* __restrict__ ea,
                                              const float* __restrict__ dis,
                                              const float* __restrict__ eew1,  // [10,4]
                                              const float* __restrict__ eeb1,  // [10]
                                              const float* __restrict__ xw1,
                                              float* __restrict__ acc, int E, int M) {
  int e = blockIdx.x;
  if (e >= M) return;
  int j = threadIdx.x;
  int row, col;
  float a0 = 0.f, a1 = 0.f, a2 = 0.f, a3 = 0.f;
  if (e < E) {
    row = ei[e]; col = ei[E + e];
    a0 = ea[e * 4 + 0]; a1 = ea[e * 4 + 1]; a2 = ea[e * 4 + 2]; a3 = ea[e * 4 + 3];
  } else {
    row = col = e - E;  // self loop, zero edge attr
  }
  float norm = dis[row] * dis[col];
  float add = 0.0f;
  if (j < 10) {
    add = eeb1[j] + a0 * eew1[j * 4 + 0] + a1 * eew1[j * 4 + 1] +
          a2 * eew1[j * 4 + 2] + a3 * eew1[j * 4 + 3];
  }
  float v = norm * (xw1[row * 128 + j] + add);
  atomicAdd(&acc[col * 128 + j], v);
}

// ---------------- relu + cvt f16 (x1h), zero acc2 ----------------
__global__ void k_relu_cvt1(const float* __restrict__ acc1, _Float16* __restrict__ xh,
                            float* __restrict__ acc2, int total) {
  int i = blockIdx.x * blockDim.x + threadIdx.x;
  if (i < total) {
    float v = fmaxf(acc1[i], 0.0f);
    xh[i] = (_Float16)v;
    acc2[i] = 0.0f;
  }
}

// ---------------- WMMA GEMM: out[N,128] = x1h[N,128] @ w2h^T + b2 ----------------
__global__ __launch_bounds__(32) void k_gemm_conv2(const _Float16* __restrict__ Ah,
                                                   const _Float16* __restrict__ Bw,  // [128,128]
                                                   const float* __restrict__ bias,
                                                   float* __restrict__ out, int N) {
  const int lane = threadIdx.x;
  const int lo = lane & 15;
  const int hi = lane >> 4;
  const int r0 = blockIdx.x * 16;

  int arow = r0 + lo;
  if (arow >= N) arow = N - 1;  // clamp, keep EXEC all-ones
  const _Float16* aptr = Ah + arow * 128 + hi * 8;

  v8f acc[8];
#pragma unroll
  for (int t = 0; t < 8; ++t) acc[t] = zero8();

#pragma unroll
  for (int ks = 0; ks < 4; ++ks) {
    const int k0 = ks * 32;
    v8h alo = *(const v8h*)(aptr + k0);
    v8h ahi = *(const v8h*)(aptr + k0 + 16);
    v16h a = pack16(alo, ahi);
#pragma unroll
    for (int nt = 0; nt < 8; ++nt) {
      const _Float16* bptr = Bw + (nt * 16 + lo) * 128 + k0 + hi * 8;
      v8h blo = *(const v8h*)(bptr);
      v8h bhi = *(const v8h*)(bptr + 16);
      v16h b = pack16(blo, bhi);
      acc[nt] = __builtin_amdgcn_wmma_f32_16x16x32_f16(
          false, a, false, b, (short)0, acc[nt], false, false);
    }
  }

#pragma unroll
  for (int nt = 0; nt < 8; ++nt) {
    const int c = nt * 16 + lo;
    const float bb = bias[c];
#pragma unroll
    for (int i = 0; i < 8; ++i) {
      int rown = r0 + i + 8 * hi;
      if (rown < N) out[rown * 128 + c] = acc[nt][i] + bb;
    }
  }
}

// ---------------- msg2: acc2[col] += norm * (xw2[row] + ee2) ----------------
__global__ __launch_bounds__(128) void k_msg2(const int* __restrict__ ei,
                                              const float* __restrict__ ea,
                                              const float* __restrict__ dis,
                                              const float* __restrict__ eew2,  // [128,4]
                                              const float* __restrict__ eeb2,  // [128]
                                              const float* __restrict__ xw2,
                                              float* __restrict__ acc, int E, int M) {
  int e = blockIdx.x;
  if (e >= M) return;
  int j = threadIdx.x;
  int row, col;
  float a0 = 0.f, a1 = 0.f, a2 = 0.f, a3 = 0.f;
  if (e < E) {
    row = ei[e]; col = ei[E + e];
    a0 = ea[e * 4 + 0]; a1 = ea[e * 4 + 1]; a2 = ea[e * 4 + 2]; a3 = ea[e * 4 + 3];
  } else {
    row = col = e - E;
  }
  float norm = dis[row] * dis[col];
  float add = eeb2[j] + a0 * eew2[j * 4 + 0] + a1 * eew2[j * 4 + 1] +
              a2 * eew2[j * 4 + 2] + a3 * eew2[j * 4 + 3];
  float v = norm * (xw2[row * 128 + j] + add);
  atomicAdd(&acc[col * 128 + j], v);
}

// ---------------- relu + cvt f16 (x2h) + partial column sums for g ----------------
__global__ __launch_bounds__(128) void k_relu_mean(const float* __restrict__ x,
                                                   _Float16* __restrict__ xh,
                                                   float* __restrict__ gsum, int N) {
  int j = threadIdx.x;
  int r0 = blockIdx.x * 64;
  int rend = r0 + 64;
  if (rend > N) rend = N;
  float s = 0.0f;
  for (int r = r0; r < rend; ++r) {
    float v = fmaxf(x[r * 128 + j], 0.0f);
    xh[r * 128 + j] = (_Float16)v;
    s += v;
  }
  atomicAdd(&gsum[j], s);
}

// ---------------- g, gcontrib (actor g-half + ab1), critic value ----------------
__global__ __launch_bounds__(128) void k_finalize(const float* __restrict__ gsum,
                                                  const float* __restrict__ aw1,  // [128,256]
                                                  const float* __restrict__ ab1,
                                                  const float* __restrict__ cw1,  // [128,128]
                                                  const float* __restrict__ cb1,
                                                  const float* __restrict__ cw2,  // [128]
                                                  const float* __restrict__ cb2,
                                                  float* __restrict__ gcontrib,
                                                  float* __restrict__ value_out, int N) {
  __shared__ float gs[128];
  __shared__ float red[128];
  int j = threadIdx.x;
  gs[j] = gsum[j] / (float)N;
  __syncthreads();
  float s = ab1[j];
  float hc = cb1[j];
#pragma unroll 4
  for (int k = 0; k < 128; ++k) {
    s += aw1[j * 256 + 128 + k] * gs[k];
    hc += cw1[j * 128 + k] * gs[k];
  }
  gcontrib[j] = s;
  hc = fmaxf(hc, 0.0f);
  red[j] = hc * cw2[j];
  __syncthreads();
  for (int off = 64; off > 0; off >>= 1) {
    if (j < off) red[j] += red[j + off];
    __syncthreads();
  }
  if (j == 0) value_out[0] = red[0] + cb2[0];
}

// ---------------- WMMA actor GEMM + fused head: logits ----------------
__global__ __launch_bounds__(32) void k_gemm_actor(const _Float16* __restrict__ Ah,   // x2h [N,128]
                                                   const _Float16* __restrict__ Bw,   // aw1h [128,256]
                                                   const float* __restrict__ gcontrib,// [128]
                                                   const float* __restrict__ aw2,    // [128]
                                                   const float* __restrict__ ab2,    // [1]
                                                   const int* __restrict__ status,
                                                   float* __restrict__ logits, int N) {
  const int lane = threadIdx.x;
  const int lo = lane & 15;
  const int hi = lane >> 4;
  const int r0 = blockIdx.x * 16;

  int arow = r0 + lo;
  if (arow >= N) arow = N - 1;
  const _Float16* aptr = Ah + arow * 128 + hi * 8;

  v8f acc[8];
#pragma unroll
  for (int t = 0; t < 8; ++t) acc[t] = zero8();

#pragma unroll
  for (int ks = 0; ks < 4; ++ks) {
    const int k0 = ks * 32;
    v8h alo = *(const v8h*)(aptr + k0);
    v8h ahi = *(const v8h*)(aptr + k0 + 16);
    v16h a = pack16(alo, ahi);
#pragma unroll
    for (int nt = 0; nt < 8; ++nt) {
      const _Float16* bptr = Bw + (nt * 16 + lo) * 256 + k0 + hi * 8;
      v8h blo = *(const v8h*)(bptr);
      v8h bhi = *(const v8h*)(bptr + 16);
      v16h b = pack16(blo, bhi);
      acc[nt] = __builtin_amdgcn_wmma_f32_16x16x32_f16(
          false, a, false, b, (short)0, acc[nt], false, false);
    }
  }

  // h = relu(acc + gcontrib[col]); partial logits p[i] = sum_col h*aw2[col]
  float p[8];
#pragma unroll
  for (int i = 0; i < 8; ++i) p[i] = 0.0f;
#pragma unroll
  for (int nt = 0; nt < 8; ++nt) {
    const int c = nt * 16 + lo;
    const float gc = gcontrib[c];
    const float w = aw2[c];
#pragma unroll
    for (int i = 0; i < 8; ++i) {
      float hv = acc[nt][i] + gc;
      hv = fmaxf(hv, 0.0f);
      p[i] += hv * w;
    }
  }
  // reduce across the 16 lanes of each half-wave (xor masks keep bit4 fixed)
#pragma unroll
  for (int m = 1; m < 16; m <<= 1) {
#pragma unroll
    for (int i = 0; i < 8; ++i) p[i] += __shfl_xor(p[i], m, 32);
  }
  if (lo == 0) {
    const float b2v = ab2[0];
#pragma unroll
    for (int i = 0; i < 8; ++i) {
      int rown = r0 + 8 * hi + i;
      if (rown < N) {
        float l = p[i] + b2v;
        l = (status[rown] == 1) ? l : -1e9f;
        logits[rown] = l;
      }
    }
  }
}

static size_t alignup(size_t x) { return (x + 255) & ~(size_t)255; }

extern "C" void kernel_launch(void* const* d_in, const int* in_sizes, int n_in,
                              void* d_out, int out_size, void* d_ws, size_t ws_size,
                              hipStream_t stream) {
  const float* node_features = (const float*)d_in[0];
  const float* edge_attr     = (const float*)d_in[1];
  const float* lin_w1        = (const float*)d_in[2];
  const float* lin_b1        = (const float*)d_in[3];
  const float* ee_w1         = (const float*)d_in[4];
  const float* ee_b1         = (const float*)d_in[5];
  const float* lin_w2        = (const float*)d_in[6];
  const float* lin_b2        = (const float*)d_in[7];
  const float* ee_w2         = (const float*)d_in[8];
  const float* ee_b2         = (const float*)d_in[9];
  const float* aw1           = (const float*)d_in[10];
  const float* ab1           = (const float*)d_in[11];
  const float* aw2           = (const float*)d_in[12];
  const float* ab2           = (const float*)d_in[13];
  const float* cw1           = (const float*)d_in[14];
  const float* cb1           = (const float*)d_in[15];
  const float* cw2           = (const float*)d_in[16];
  const float* cb2           = (const float*)d_in[17];
  const int*   edge_index    = (const int*)d_in[18];
  const int*   status        = (const int*)d_in[19];

  const int N = in_sizes[19];       // 50000
  const int E = in_sizes[18] / 2;   // 600000
  const int M = E + N;              // edges + self loops

  float* out = (float*)d_out;       // [0..N) logits, [N] value

  // ---- workspace carve ----
  char* p = (char*)d_ws;
  float*    A    = (float*)p;    p += alignup((size_t)N * 128 * 4);  // xw1 -> acc2 -> x2
  float*    Bacc = (float*)p;    p += alignup((size_t)N * 128 * 4);  // acc1 -> xw2
  _Float16* Hh   = (_Float16*)p; p += alignup((size_t)N * 128 * 2);  // x1h -> x2h
  float*    dis  = (float*)p;    p += alignup((size_t)N * 4);        // deg -> rsqrt(deg)
  _Float16* w2h  = (_Float16*)p; p += alignup((size_t)128 * 128 * 2);
  _Float16* aw1h = (_Float16*)p; p += alignup((size_t)128 * 256 * 2);
  float*    gsum = (float*)p;    p += alignup((size_t)128 * 4);
  float*    gctr = (float*)p;    p += alignup((size_t)128 * 4);

  // 1. init acc1=0, deg=1, gsum=0
  k_init<<<4096, 256, 0, stream>>>(Bacc, dis, gsum, N);
  // 2. degree atomics over rows
  k_degree<<<(E + 255) / 256, 256, 0, stream>>>(edge_index, dis, E);
  // 3. dis = rsqrt(deg)
  k_rsqrt<<<(N + 255) / 256, 256, 0, stream>>>(dis, N);
  // 4. convert weights to f16
  k_cvtw<<<(128 * 256 + 255) / 256, 256, 0, stream>>>(lin_w2, aw1, w2h, aw1h);
  // 5. conv1 dense: xw1 -> A
  k_xw1<<<N, 128, 0, stream>>>(node_features, lin_w1, lin_b1, A, N);
  // 6. msg1: scatter into acc1 (Bacc)
  k_msg1<<<M, 128, 0, stream>>>(edge_index, edge_attr, dis, ee_w1, ee_b1, A, Bacc, E, M);
  // 7. x1h = f16(relu(acc1)); zero acc2 (A)
  k_relu_cvt1<<<(N * 128 + 255) / 256, 256, 0, stream>>>(Bacc, Hh, A, N * 128);
  // 8. WMMA conv2 GEMM: xw2 (Bacc) = x1h @ w2^T + b2
  k_gemm_conv2<<<(N + 15) / 16, 32, 0, stream>>>(Hh, w2h, lin_b2, Bacc, N);
  // 9. msg2: scatter into acc2 (A)
  k_msg2<<<M, 128, 0, stream>>>(edge_index, edge_attr, dis, ee_w2, ee_b2, Bacc, A, E, M);
  // 10. x2h = f16(relu(acc2)); column partial sums -> gsum
  k_relu_mean<<<(N + 63) / 64, 128, 0, stream>>>(A, Hh, gsum, N);
  // 11. g, gcontrib, critic value -> out[N]
  k_finalize<<<1, 128, 0, stream>>>(gsum, aw1, ab1, cw1, cb1, cw2, cb2, gctr, out + N, N);
  // 12. WMMA actor GEMM + fused head -> logits
  k_gemm_actor<<<(N + 15) / 16, 32, 0, stream>>>(Hh, aw1h, gctr, aw2, ab2, status, out, N);
}
